// HNN_68496138437411
// MI455X (gfx1250) — compile-verified, matrix operand order checked by reading, not run on hardware
//
#include <hip/hip_runtime.h>
#include <hip/hip_bf16.h>

// ---------------------------------------------------------------------------
// Sparse hierarchical NN, fused pipeline for MI455X (gfx1250, wave32).
//
// Roofline: 3.9 GFLOP total, ~350 MB layered HBM traffic -> ~15us @ 23.3TB/s.
// Strategy: transpose to feature-major [L,B] so CSR sparse layers do
// scalar-uniform (s_load) edge reads + fully coalesced float4 batch gathers
// served mostly from the 192MB L2. fc heads use v_wmma_f32_16x16x32_f16 with
// branchless operand construction (no exec-mask churn in the K loop).
// ---------------------------------------------------------------------------

#define BDIM 8192
#define L1D  4096
#define L2D  2048
#define L3D  1024
#define L4D  512
#define NNZ1 131072
#define NNZ2 65536
#define NNZ3 32768

typedef __attribute__((ext_vector_type(16))) _Float16 v16h;
typedef __attribute__((ext_vector_type(8)))  float    v8f;

// ---------------- transpose [rows,cols] -> [cols,rows] ----------------------
__global__ void k_transpose(const float* __restrict__ in, float* __restrict__ out,
                            int rows, int cols) {
  __shared__ float tile[32][33];
  const int bx = blockIdx.x * 32;           // col base
  const int by = blockIdx.y * 32;           // row base
  const int tx = threadIdx.x, ty = threadIdx.y;   // block (32, 8)
#pragma unroll
  for (int j = 0; j < 32; j += 8)
    tile[ty + j][tx] = in[(size_t)(by + ty + j) * cols + (bx + tx)];
  __syncthreads();
#pragma unroll
  for (int j = 0; j < 32; j += 8)
    out[(size_t)(bx + ty + j) * rows + (by + tx)] = tile[tx][ty + j];
}

// ---------------- CSR build: zero / histogram / scan / scatter --------------
__global__ void k_zero_i32(int* __restrict__ p, int n) {
  int i = blockIdx.x * 256 + threadIdx.x;
  if (i < n) p[i] = 0;
}

__global__ void k_hist(const int* __restrict__ out_idx, int* __restrict__ counts, int nnz) {
  int e = blockIdx.x * 256 + threadIdx.x;
  if (e < nnz) atomicAdd(&counts[out_idx[e]], 1);
}

// single block, 256 threads; n is a multiple of 256; offsets has n+1 entries
__global__ void k_scan(const int* __restrict__ counts, int* __restrict__ offsets,
                       int* __restrict__ cursor, int n) {
  __shared__ int part[256];
  const int tid = threadIdx.x;
  const int chunk = n / 256;
  const int lo = tid * chunk, hi = lo + chunk;
  int s = 0;
  for (int i = lo; i < hi; ++i) s += counts[i];
  part[tid] = s;
  __syncthreads();
  if (tid == 0) {
    int run = 0;
    for (int i = 0; i < 256; ++i) { int t = part[i]; part[i] = run; run += t; }
  }
  __syncthreads();
  int run = part[tid];
  for (int i = lo; i < hi; ++i) {
    offsets[i] = run; cursor[i] = run; run += counts[i];
  }
  if (hi == n) offsets[n] = run;   // thread 255 writes total nnz
}

__global__ void k_scatter(const float* __restrict__ w, const int* __restrict__ out_idx,
                          const int* __restrict__ in_idx, int* __restrict__ cursor,
                          float* __restrict__ csr_w, int* __restrict__ csr_in, int nnz) {
  int e = blockIdx.x * 256 + threadIdx.x;
  if (e < nnz) {
    int o = out_idx[e];
    int p = atomicAdd(&cursor[o], 1);
    csr_w[p]  = w[e];
    csr_in[p] = in_idx[e];
  }
}

// ---------------- sparse layer: outT[o,b] = relu(bias[o] + sum w*inT[in,b]) -
// grid: (BDIM/1024, n_out), block 256. Each thread handles 4 batch elems.
// csr_w/csr_in reads are wave-uniform -> scalar loads; gathers are coalesced
// float4. __builtin_prefetch emits global_prefetch_b8 (CDNA5) with a
// branchless clamped lookahead to hide the index->gather latency chain.
__global__ void k_sparse_csr(const float* __restrict__ inT, float* __restrict__ outT,
                             const float* __restrict__ csr_w, const int* __restrict__ csr_in,
                             const int* __restrict__ offsets, const float* __restrict__ bias) {
  const int o = blockIdx.y;
  const int b = (blockIdx.x * 256 + threadIdx.x) * 4;
  const float bv = bias[o];
  float4 acc = make_float4(bv, bv, bv, bv);
  const int e0 = offsets[o], e1 = offsets[o + 1];
  for (int e = e0; e < e1; ++e) {
    const float w  = csr_w[e];
    const int  idx = csr_in[e];
    const int  pe  = (e + 4 < e1) ? (e + 4) : (e1 - 1);      // branchless lookahead
    __builtin_prefetch(&inT[(size_t)csr_in[pe] * BDIM + b], 0, 1);
    const float4 v = *(const float4*)&inT[(size_t)idx * BDIM + b];
    acc.x = fmaf(w, v.x, acc.x);
    acc.y = fmaf(w, v.y, acc.y);
    acc.z = fmaf(w, v.z, acc.z);
    acc.w = fmaf(w, v.w, acc.w);
  }
  float4 r = make_float4(fmaxf(acc.x, 0.f), fmaxf(acc.y, 0.f),
                         fmaxf(acc.z, 0.f), fmaxf(acc.w, 0.f));
  *(float4*)&outT[(size_t)o * BDIM + b] = r;
}

// ---------------- fc head via WMMA: fout[b] = relu(dot(w, inT[:,b]) + bias) -
// A (16x32 f16): row 0 = weight chunk (masked per-lane, NO exec branching),
//                rows 1..15 = 0.
// B (32x16 f16): 16 batch columns of inT.
// D row 0 (lanes 0..15, c[0]) = the 16 dot products.
// block 256 = 8 waves, each wave covers 16 batch columns.
__global__ void k_fc_wmma(const float* __restrict__ inT, const float* __restrict__ w,
                          const float* __restrict__ bias, float* __restrict__ fout,
                          int K) {
  const int tid  = threadIdx.x;
  const int wave = tid >> 5;
  const int lane = tid & 31;
  const int n    = lane & 15;
  const int hi   = lane >> 4;
  const int b0   = (blockIdx.x * (blockDim.x >> 5) + wave) << 4;
  const int col  = b0 + n;
  const float msk = (n == 0) ? 1.0f : 0.0f;   // row-0 lane mask (v_cndmask)

  v8f c = {};
  for (int k0 = 0; k0 < K; k0 += 32) {
    const int kb1 = k0 + hi * 8;        // K layout: lanes 0-15 {0..7,16..23},
    const int kb2 = kb1 + 16;           //           lanes 16-31 {8..15,24..31}
    v16h a, bm;
#pragma unroll
    for (int t = 0; t < 8; ++t) {
      // unconditional loads (broadcast within each half-wave) + VALU mask
      a[t]     = (_Float16)(msk * w[kb1 + t]);
      a[t + 8] = (_Float16)(msk * w[kb2 + t]);
      bm[t]     = (_Float16)inT[(size_t)(kb1 + t) * BDIM + col];
      bm[t + 8] = (_Float16)inT[(size_t)(kb2 + t) * BDIM + col];
    }
    c = __builtin_amdgcn_wmma_f32_16x16x32_f16(false, a, false, bm,
                                               (short)0, c, false, false);
  }
  if (lane < 16)                        // D row M=0 lives in c[0], lanes 0..15
    fout[b0 + n] = fmaxf(c[0] + bias[0], 0.f);
}

// ---------------- final combine: out = ro_b + sum ro_w[i]*f_i --------------
__global__ void k_combine(const float* __restrict__ f1, const float* __restrict__ f2,
                          const float* __restrict__ f3, const float* __restrict__ f4,
                          const float* __restrict__ ro_w, const float* __restrict__ ro_b,
                          float* __restrict__ out) {
  int b = blockIdx.x * 256 + threadIdx.x;
  out[b] = ro_b[0] + ro_w[0] * f1[b] + ro_w[1] * f2[b]
                   + ro_w[2] * f3[b] + ro_w[3] * f4[b];
}

// ---------------------------------------------------------------------------
extern "C" void kernel_launch(void* const* d_in, const int* in_sizes, int n_in,
                              void* d_out, int out_size, void* d_ws, size_t ws_size,
                              hipStream_t stream) {
  (void)in_sizes; (void)n_in; (void)out_size; (void)ws_size;

  const float* x      = (const float*)d_in[0];
  const float* sl1_w  = (const float*)d_in[1];
  const float* sl1_b  = (const float*)d_in[2];
  const int*   sl1_o  = (const int*)  d_in[3];
  const int*   sl1_i  = (const int*)  d_in[4];
  const float* fc1_w  = (const float*)d_in[5];
  const float* fc1_b  = (const float*)d_in[6];
  const float* sl2_w  = (const float*)d_in[7];
  const float* sl2_b  = (const float*)d_in[8];
  const int*   sl2_o  = (const int*)  d_in[9];
  const int*   sl2_i  = (const int*)  d_in[10];
  const float* fc2_w  = (const float*)d_in[11];
  const float* fc2_b  = (const float*)d_in[12];
  const float* sl3_w  = (const float*)d_in[13];
  const float* sl3_b  = (const float*)d_in[14];
  const int*   sl3_o  = (const int*)  d_in[15];
  const int*   sl3_i  = (const int*)  d_in[16];
  const float* fc3_w  = (const float*)d_in[17];
  const float* fc3_b  = (const float*)d_in[18];
  const float* fc4_w  = (const float*)d_in[19];
  const float* fc4_b  = (const float*)d_in[20];
  const float* ro_w   = (const float*)d_in[21];
  const float* ro_b   = (const float*)d_in[22];
  float* out = (float*)d_out;

  // ---- workspace carve-up (256B aligned) ----
  char* ws = (char*)d_ws;
  size_t off = 0;
  auto take = [&](size_t bytes) -> char* {
    char* p = ws + off;
    off = (off + bytes + 255) & ~(size_t)255;
    return p;
  };
  float* xT   = (float*)take((size_t)L1D * BDIM * 4);
  float* s1T  = (float*)take((size_t)L2D * BDIM * 4);
  float* s2T  = (float*)take((size_t)L3D * BDIM * 4);
  float* s3T  = (float*)take((size_t)L4D * BDIM * 4);
  float* f1   = (float*)take((size_t)BDIM * 4);
  float* f2   = (float*)take((size_t)BDIM * 4);
  float* f3   = (float*)take((size_t)BDIM * 4);
  float* f4   = (float*)take((size_t)BDIM * 4);
  float* c1w  = (float*)take((size_t)NNZ1 * 4);
  int*   c1i  = (int*)  take((size_t)NNZ1 * 4);
  float* c2w  = (float*)take((size_t)NNZ2 * 4);
  int*   c2i  = (int*)  take((size_t)NNZ2 * 4);
  float* c3w  = (float*)take((size_t)NNZ3 * 4);
  int*   c3i  = (int*)  take((size_t)NNZ3 * 4);
  int* cnt1 = (int*)take((L2D + 1) * 4);
  int* off1 = (int*)take((L2D + 1) * 4);
  int* cur1 = (int*)take((L2D + 1) * 4);
  int* cnt2 = (int*)take((L3D + 1) * 4);
  int* off2 = (int*)take((L3D + 1) * 4);
  int* cur2 = (int*)take((L3D + 1) * 4);
  int* cnt3 = (int*)take((L4D + 1) * 4);
  int* off3 = (int*)take((L4D + 1) * 4);
  int* cur3 = (int*)take((L4D + 1) * 4);

  // ---- CSR build (3 layers) ----
  k_zero_i32<<<(L2D + 256) / 256, 256, 0, stream>>>(cnt1, L2D + 1);
  k_zero_i32<<<(L3D + 256) / 256, 256, 0, stream>>>(cnt2, L3D + 1);
  k_zero_i32<<<(L4D + 256) / 256, 256, 0, stream>>>(cnt3, L4D + 1);
  k_hist<<<NNZ1 / 256, 256, 0, stream>>>(sl1_o, cnt1, NNZ1);
  k_hist<<<NNZ2 / 256, 256, 0, stream>>>(sl2_o, cnt2, NNZ2);
  k_hist<<<NNZ3 / 256, 256, 0, stream>>>(sl3_o, cnt3, NNZ3);
  k_scan<<<1, 256, 0, stream>>>(cnt1, off1, cur1, L2D);
  k_scan<<<1, 256, 0, stream>>>(cnt2, off2, cur2, L3D);
  k_scan<<<1, 256, 0, stream>>>(cnt3, off3, cur3, L4D);
  k_scatter<<<NNZ1 / 256, 256, 0, stream>>>(sl1_w, sl1_o, sl1_i, cur1, c1w, c1i, NNZ1);
  k_scatter<<<NNZ2 / 256, 256, 0, stream>>>(sl2_w, sl2_o, sl2_i, cur2, c2w, c2i, NNZ2);
  k_scatter<<<NNZ3 / 256, 256, 0, stream>>>(sl3_w, sl3_o, sl3_i, cur3, c3w, c3i, NNZ3);

  // ---- transpose x: [B, L1] -> [L1, B] ----
  {
    dim3 g(L1D / 32, BDIM / 32), blk(32, 8);
    k_transpose<<<g, blk, 0, stream>>>(x, xT, BDIM, L1D);
  }

  const int FC_BLOCKS = BDIM / 128;  // 8 waves/block * 16 cols/wave

  // ---- pipeline ----
  k_fc_wmma<<<FC_BLOCKS, 256, 0, stream>>>(xT, fc1_w, fc1_b, f1, L1D);
  {
    dim3 g(BDIM / 1024, L2D);
    k_sparse_csr<<<g, 256, 0, stream>>>(xT, s1T, c1w, c1i, off1, sl1_b);
  }
  k_fc_wmma<<<FC_BLOCKS, 256, 0, stream>>>(s1T, fc2_w, fc2_b, f2, L2D);
  {
    dim3 g(BDIM / 1024, L3D);
    k_sparse_csr<<<g, 256, 0, stream>>>(s1T, s2T, c2w, c2i, off2, sl2_b);
  }
  k_fc_wmma<<<FC_BLOCKS, 256, 0, stream>>>(s2T, fc3_w, fc3_b, f3, L3D);
  {
    dim3 g(BDIM / 1024, L4D);
    k_sparse_csr<<<g, 256, 0, stream>>>(s2T, s3T, c3w, c3i, off3, sl3_b);
  }
  k_fc_wmma<<<FC_BLOCKS, 256, 0, stream>>>(s3T, fc4_w, fc4_b, f4, L4D);

  k_combine<<<BDIM / 256, 256, 0, stream>>>(f1, f2, f3, f4, ro_w, ro_b, out);
}
